// DetectHead_15839839387766
// MI455X (gfx1250) — matrix operand order repaired and checked by reading, not compiled, not run on hardware
//
#include <hip/hip_runtime.h>
#include <hip/hip_bf16.h>

// ---------------------------------------------------------------------------
// MI455X (gfx1250) detect-head. Compute-bound (~95 GFLOP, L2-resident data),
// convs run as implicit GEMMs on v_wmma_f32_16x16x32_bf16 (fp32 accumulate).
// K order = (ky*3+kx)*C + ci so each 32-wide K chunk sits inside one 3x3 tap:
// tap bounds-check + row pointer are per-tap scalars, and A fragments are two
// contiguous b128 loads from an NHWC-bf16 copy of the features.
// ---------------------------------------------------------------------------

typedef __attribute__((ext_vector_type(16))) __bf16 v16bf;
typedef __attribute__((ext_vector_type(8)))  __bf16 v8bf;
typedef __attribute__((ext_vector_type(8)))  float  v8f;

#define NUM_CLS 80
#define REG_CH  68   // 4 * 17
#define KREG    17
#define NPAD    80   // reg N padded to 5 full 16-wide tiles
#define BATCH_N 8

__device__ __forceinline__ unsigned short bf_bits(float f) {
    unsigned u = __float_as_uint(f);
    return (unsigned short)((u + 0x7FFFu + ((u >> 16) & 1u)) >> 16); // RNE
}

// ---------------- preparation kernels ---------------------------------------

__global__ void zero_fill_kernel(unsigned short* __restrict__ p, int n) {
    int i = blockIdx.x * blockDim.x + threadIdx.x;
    if (i < n) p[i] = 0;
}

// NCHW fp32 -> NHWC bf16 : out[(b*HW+pos)*C + c] = in[(b*C+c)*HW + pos]
__global__ void nchw_to_nhwc_bf16(const float* __restrict__ in,
                                  unsigned short* __restrict__ out,
                                  int C, int HW, int total) {
    int i = blockIdx.x * blockDim.x + threadIdx.x;
    if (i >= total) return;
    int m = i / C, c = i - m * C;
    int b = m / HW, pos = m - b * HW;
    out[i] = bf_bits(in[((size_t)b * C + c) * HW + pos]);
}

// w1 OIHW (co,ci,ky,kx) -> bf16 [co][ (ky*3+kx)*C + ci ]
__global__ void pack_w1_kernel(const float* __restrict__ w,
                               unsigned short* __restrict__ out, int C, int total) {
    int i = blockIdx.x * blockDim.x + threadIdx.x;
    if (i >= total) return;
    int co = i / (C * 9);
    int rem = i - co * C * 9;
    int r = rem / C, ci = rem - r * C;
    out[i] = bf_bits(w[(size_t)co * C * 9 + (size_t)ci * 9 + r]);
}

// w2 (Nout,C) -> bf16 [NPAD][C], zero rows beyond Nout
__global__ void pack_w2_kernel(const float* __restrict__ w2,
                               unsigned short* __restrict__ out, int C, int Nout) {
    int i = blockIdx.x * blockDim.x + threadIdx.x;
    if (i < NPAD * C) {
        int nn = i / C, cc = i - nn * C;
        out[i] = (nn < Nout) ? bf_bits(w2[(size_t)nn * C + cc]) : (unsigned short)0;
    }
}

// scale = g/sqrt(v+eps); bias = beta - mean*scale
__global__ void bn_prep_kernel(const float* __restrict__ g, const float* __restrict__ be,
                               const float* __restrict__ mu, const float* __restrict__ va,
                               float* __restrict__ a, float* __restrict__ b, int C) {
    int i = blockIdx.x * blockDim.x + threadIdx.x;
    if (i < C) {
        float s = g[i] * rsqrtf(va[i] + 1e-5f);
        a[i] = s;
        b[i] = be[i] - mu[i] * s;
    }
}

// ---------------- fused 3x3 conv + BN + SiLU (implicit GEMM, WMMA bf16) -----
// M = B*H*W, N = C, K = 9*C. Each wave owns a 16(M) x 32(N) tile (A reused by
// two WMMAs). Grid: (M/128, C/32), 256 threads = 8 waves.
template<int C, int HH>
__global__ __launch_bounds__(256)
void conv3x3_bn_silu_wmma(const unsigned short* __restrict__ featbf, // bf16 [B*HW][C]
                          const unsigned short* __restrict__ wpk,    // bf16 [C][9*C]
                          const unsigned short* __restrict__ zrow,   // >= C+32 zeros
                          const float* __restrict__ bna,
                          const float* __restrict__ bnb,
                          unsigned short* __restrict__ act) {        // bf16 [M][C]
    constexpr int HW = HH * HH;
    constexpr int K9 = 9 * C;
    const int lane = threadIdx.x & 31;
    const int wave = threadIdx.x >> 5;
    const int m0 = (blockIdx.x * 8 + wave) * 16;
    const int n0 = blockIdx.y * 32;
    const int lh = lane & 15;
    const int khalf8  = (lane >> 4) * 16;  // A-layout K sub-block (contiguous, matches B)
    const int khalf16 = (lane >> 4) * 16;  // B-layout K sub-block

    const int mrow = m0 + lh;
    const int b   = mrow / HW;
    const int pos = mrow - b * HW;
    const int oy  = pos / HH, ox = pos - oy * HH;

    const unsigned short* bp0 = wpk + (size_t)(n0 + lh) * K9 + khalf16;
    const unsigned short* bp1 = bp0 + (size_t)16 * K9;
    const unsigned short* zp  = zrow + khalf8;

    v8f acc0 = {}, acc1 = {};
#pragma unroll
    for (int r = 0; r < 9; ++r) {
        const int ky = r / 3, kx = r - ky * 3;      // compile-time (unrolled)
        const int iy = oy + ky - 1, ix = ox + kx - 1;
        const bool inb = ((unsigned)iy < (unsigned)HH) & ((unsigned)ix < (unsigned)HH);
        const unsigned short* ap =
            inb ? (featbf + ((size_t)((b * HH + iy) * HH + ix)) * C + khalf8) : zp;
        const int kbase = r * C;
#pragma unroll 2
        for (int ci0 = 0; ci0 < C; ci0 += 32) {
            v8bf lo = *(const v8bf*)(ap + ci0);
            v8bf hi = *(const v8bf*)(ap + ci0 + 8);
            v16bf a = __builtin_shufflevector(lo, hi,
                0, 1, 2, 3, 4, 5, 6, 7, 8, 9, 10, 11, 12, 13, 14, 15);
            v16bf bf0 = *(const v16bf*)(bp0 + kbase + ci0);
            v16bf bf1 = *(const v16bf*)(bp1 + kbase + ci0);
            acc0 = __builtin_amdgcn_wmma_f32_16x16x32_bf16(
                false, a, false, bf0, (short)0, acc0, false, false);
            acc1 = __builtin_amdgcn_wmma_f32_16x16x32_bf16(
                false, a, false, bf1, (short)0, acc1, false, false);
        }
    }

    // Epilogue: C/D layout — lane gives N col, vgpr r gives M (lane-half +8).
    const int na = n0 + lh, nb = n0 + 16 + lh;
    const float sa0 = bna[na], sb0 = bnb[na];
    const float sa1 = bna[nb], sb1 = bnb[nb];
    const int mb = m0 + (lane >> 4) * 8;
#pragma unroll
    for (int r = 0; r < 8; ++r) {
        float x0 = acc0[r] * sa0 + sb0;
        float x1 = acc1[r] * sa1 + sb1;
        float y0 = x0 * (1.f / (1.f + __expf(-x0)));   // SiLU
        float y1 = x1 * (1.f / (1.f + __expf(-x1)));
        act[(size_t)(mb + r) * C + na] = bf_bits(y0);
        act[(size_t)(mb + r) * C + nb] = bf_bits(y1);
    }
}

// ---------------- 1x1 conv head (GEMM M x Nout x C, WMMA bf16) --------------
template<int C>
__global__ __launch_bounds__(256)
void conv1x1_wmma(const unsigned short* __restrict__ act,   // bf16 [M][C]
                  const unsigned short* __restrict__ wpk2,  // bf16 [NPAD][C]
                  const float* __restrict__ bias,           // [Nout]
                  float* __restrict__ out,                  // fp32 [M][Nout]
                  int Nout) {
    const int lane = threadIdx.x & 31;
    const int wave = threadIdx.x >> 5;
    const int m0 = (blockIdx.x * 8 + wave) * 16;
    const int n0 = blockIdx.y * 16;
    const int lh = lane & 15;
    const int khalf8 = (lane >> 4) * 16;

    const unsigned short* ap = act + (size_t)(m0 + lh) * C + khalf8;
    const unsigned short* bp = wpk2 + (size_t)(n0 + lh) * C + (lane >> 4) * 16;

    v8f acc = {};
#pragma unroll
    for (int kt = 0; kt < C; kt += 32) {
        v8bf lo = *(const v8bf*)(ap + kt);
        v8bf hi = *(const v8bf*)(ap + kt + 8);
        v16bf a = __builtin_shufflevector(lo, hi,
            0, 1, 2, 3, 4, 5, 6, 7, 8, 9, 10, 11, 12, 13, 14, 15);
        v16bf bfrag = *(const v16bf*)(bp + kt);
        acc = __builtin_amdgcn_wmma_f32_16x16x32_bf16(
            false, a, false, bfrag, (short)0, acc, false, false);
    }

    const int n = n0 + lh;
    if (n < Nout) {
        const float bs = bias[n];
        const int mb = m0 + (lane >> 4) * 8;
#pragma unroll
        for (int r = 0; r < 8; ++r)
            out[(size_t)(mb + r) * Nout + n] = acc[r] + bs;
    }
}

// ---------------- decode: sigmoid/argmax + DFL softmax + grid boxes ---------
__global__ void decode_kernel(const float* __restrict__ cls,  // [M][80]
                              const float* __restrict__ reg,  // [M][68]
                              float* __restrict__ boxesU, float* __restrict__ scoresU,
                              int* __restrict__ labelsU, int* __restrict__ validU,
                              int HWsz, int W, float stride, int Mtot) {
    int m = blockIdx.x * blockDim.x + threadIdx.x;
    if (m >= Mtot) return;

    const float* cr = cls + (size_t)m * NUM_CLS;
    float best = cr[0]; int bi = 0;
    for (int c = 1; c < NUM_CLS; ++c) {
        float v = cr[c];
        if (v > best) { best = v; bi = c; }   // argmax(logits)==argmax(sigmoid)
    }
    float score = 1.f / (1.f + __expf(-best));

    const float* rr = reg + (size_t)m * REG_CH;
    float d[4];
#pragma unroll
    for (int s = 0; s < 4; ++s) {
        const float* p = rr + s * KREG;
        float mx = p[0];
        for (int j = 1; j < KREG; ++j) mx = fmaxf(mx, p[j]);
        float se = 0.f, sw = 0.f;
        for (int j = 0; j < KREG; ++j) {
            float e = __expf(p[j] - mx);
            se += e; sw += e * (float)j;
        }
        d[s] = (sw / se) * stride;
    }
    int pos = m % HWsz;
    int oy = pos / W, ox = pos - oy * W;
    float gx = (ox + 0.5f) * stride, gy = (oy + 0.5f) * stride;
    float4 bx = make_float4(gx - d[0], gy - d[1], gx + d[2], gy + d[3]);
    ((float4*)boxesU)[m] = bx;
    scoresU[m] = score;
    labelsU[m] = bi;
    validU[m]  = (score > 0.25f) ? 1 : 0;
}

// ---------------- stable descending argsort via rank counting ---------------
// rank(i) = #{j : s_j > s_i or (s_j == s_i and j < i)}  == stable argsort(-s).
__global__ __launch_bounds__(256)
void argsort_scatter_kernel(const float* __restrict__ boxesU, const float* __restrict__ scoresU,
                            const int* __restrict__ labelsU, const int* __restrict__ validU,
                            float* __restrict__ oBoxes, float* __restrict__ oScores,
                            float* __restrict__ oLabels,
                            int* __restrict__ sLabels, int* __restrict__ sValid, int N) {
    const int b   = blockIdx.y;
    const int i   = blockIdx.x * blockDim.x + threadIdx.x;
    const int tid = threadIdx.x;
    __shared__ float ss[256];

    float my = (i < N) ? scoresU[(size_t)b * N + i] : 0.f;
    int rank = 0;
    for (int base = 0; base < N; base += 256) {
        int j = base + tid;
        ss[tid] = (j < N) ? scoresU[(size_t)b * N + j] : -3.0e38f;
        __syncthreads();
        int lim = N - base; if (lim > 256) lim = 256;
        for (int t = 0; t < lim; ++t) {
            float s = ss[t]; int jj = base + t;
            rank += (s > my) || (s == my && jj < i);
        }
        __syncthreads();
    }
    if (i < N) {
        size_t src = (size_t)b * N + i, dst = (size_t)b * N + rank;
        ((float4*)oBoxes)[dst] = ((const float4*)boxesU)[src];
        oScores[dst] = my;
        oLabels[dst] = (float)labelsU[src];
        sLabels[dst] = labelsU[src];
        sValid[dst]  = validU[src];
    }
}

// ---------------- greedy per-class NMS (one block per image) ----------------
__global__ __launch_bounds__(256)
void nms_kernel(const float* __restrict__ boxes,  // sorted [B,N,4]
                const int* __restrict__ labels, const int* __restrict__ valid,
                float* __restrict__ keep, int N) {
    const int b = blockIdx.x, tid = threadIdx.x;
    __shared__ unsigned int supp[200];            // up to 6400 bits
    __shared__ float bx1, by1, bx2, by2, barea;
    __shared__ int blab, bact;

    for (int w = tid; w < 200; w += 256) supp[w] = 0u;
    __syncthreads();

    const float* bb = boxes + (size_t)b * N * 4;
    const int*   lb = labels + (size_t)b * N;

    for (int i = 0; i < N; ++i) {
        if (tid == 0) {
            int a = valid[(size_t)b * N + i] && !((supp[i >> 5] >> (i & 31)) & 1u);
            bact = a;
            if (a) {
                bx1 = bb[i * 4 + 0]; by1 = bb[i * 4 + 1];
                bx2 = bb[i * 4 + 2]; by2 = bb[i * 4 + 3];
                barea = (bx2 - bx1) * (by2 - by1);
                blab = lb[i];
            }
        }
        __syncthreads();
        if (bact) {
            for (int j = i + 1 + tid; j < N; j += 256) {
                if (lb[j] == blab && !((supp[j >> 5] >> (j & 31)) & 1u)) {
                    float x1 = bb[j * 4 + 0], y1 = bb[j * 4 + 1];
                    float x2 = bb[j * 4 + 2], y2 = bb[j * 4 + 3];
                    float iw = fminf(bx2, x2) - fmaxf(bx1, x1);
                    float ih = fminf(by2, y2) - fmaxf(by1, y1);
                    float inter = fmaxf(iw, 0.f) * fmaxf(ih, 0.f);
                    float area = (x2 - x1) * (y2 - y1);
                    float iou = inter / (barea + area - inter + 1e-6f);
                    if (iou > 0.45f) atomicOr(&supp[j >> 5], 1u << (j & 31));
                }
            }
        }
        __syncthreads();
    }
    for (int j = tid; j < N; j += 256)
        keep[(size_t)b * N + j] =
            (valid[(size_t)b * N + j] && !((supp[j >> 5] >> (j & 31)) & 1u)) ? 1.f : 0.f;
}

// ---------------------------------------------------------------------------

extern "C" void kernel_launch(void* const* d_in, const int* in_sizes, int n_in,
                              void* d_out, int out_size, void* d_ws, size_t ws_size,
                              hipStream_t stream) {
    (void)in_sizes; (void)n_in; (void)out_size; (void)ws_size;

    const int   Cs[3]  = {128, 256, 512};
    const int   Ss[3]  = {80, 40, 20};
    const float Sv[3]  = {8.f, 16.f, 32.f};

    const float* feats[3] = {(const float*)d_in[0], (const float*)d_in[1], (const float*)d_in[2]};
    // params per level: 14 tensors; per branch (cls=0, reg=1) 7 tensors:
    // 0=w1, 1=bn_g, 2=bn_b, 3=bn_m, 4=bn_v, 5=w2, 6=b2
    auto P = [&](int L, int br, int k) -> const float* {
        return (const float*)d_in[3 + L * 14 + br * 7 + k];
    };

    char* ws = (char*)d_ws;
    size_t off = 0;
    auto alloc = [&](size_t bytes) -> char* {
        char* p = ws + off;
        off = (off + bytes + 255) & ~(size_t)255;
        return p;
    };

    unsigned short *w1pk[3][2], *w2pk[3][2], *actb[3][2], *featbf[3], *zrow;
    float *bna[3][2], *bnb[3][2], *hout[3][2];
    float *boxesU[3], *scoresU[3];
    int *labelsU[3], *validU[3], *sLabels[3], *sValid[3];

    zrow = (unsigned short*)alloc(1088 * 2);   // >= Cmax+32 zeros
    for (int L = 0; L < 3; ++L) {
        const int C = Cs[L], H = Ss[L];
        const size_t M = (size_t)BATCH_N * H * H;
        featbf[L] = (unsigned short*)alloc(M * C * 2);
        for (int br = 0; br < 2; ++br) {
            const int Nout = br == 0 ? NUM_CLS : REG_CH;
            w1pk[L][br] = (unsigned short*)alloc((size_t)C * C * 9 * 2);
            w2pk[L][br] = (unsigned short*)alloc((size_t)NPAD * C * 2);
            bna[L][br]  = (float*)alloc((size_t)C * 4);
            bnb[L][br]  = (float*)alloc((size_t)C * 4);
            actb[L][br] = (unsigned short*)alloc(M * C * 2);
            hout[L][br] = (float*)alloc(M * Nout * 4);
        }
        boxesU[L]  = (float*)alloc(M * 4 * 4);
        scoresU[L] = (float*)alloc(M * 4);
        labelsU[L] = (int*)alloc(M * 4);
        validU[L]  = (int*)alloc(M * 4);
        sLabels[L] = (int*)alloc(M * 4);
        sValid[L]  = (int*)alloc(M * 4);
    }

    zero_fill_kernel<<<(1088 + 255) / 256, 256, 0, stream>>>(zrow, 1088);

    // ---- prep + GEMMs ----
    for (int L = 0; L < 3; ++L) {
        const int C = Cs[L], H = Ss[L];
        const int HW = H * H;
        const int M = BATCH_N * HW;

        nchw_to_nhwc_bf16<<<((size_t)M * C + 255) / 256, 256, 0, stream>>>(
            feats[L], featbf[L], C, HW, M * C);

        for (int br = 0; br < 2; ++br) {
            const int Nout = br == 0 ? NUM_CLS : REG_CH;
            const int nw1 = C * C * 9;
            bn_prep_kernel<<<(C + 255) / 256, 256, 0, stream>>>(
                P(L, br, 1), P(L, br, 2), P(L, br, 3), P(L, br, 4),
                bna[L][br], bnb[L][br], C);
            pack_w1_kernel<<<(nw1 + 255) / 256, 256, 0, stream>>>(
                P(L, br, 0), w1pk[L][br], C, nw1);
            pack_w2_kernel<<<(NPAD * C + 255) / 256, 256, 0, stream>>>(
                P(L, br, 5), w2pk[L][br], C, Nout);

            dim3 g3(M / 128, C / 32);
            dim3 g1(M / 128, NPAD / 16);
            switch (L) {
            case 0:
                conv3x3_bn_silu_wmma<128, 80><<<g3, 256, 0, stream>>>(
                    featbf[L], w1pk[L][br], zrow, bna[L][br], bnb[L][br], actb[L][br]);
                conv1x1_wmma<128><<<g1, 256, 0, stream>>>(
                    actb[L][br], w2pk[L][br], P(L, br, 6), hout[L][br], Nout);
                break;
            case 1:
                conv3x3_bn_silu_wmma<256, 40><<<g3, 256, 0, stream>>>(
                    featbf[L], w1pk[L][br], zrow, bna[L][br], bnb[L][br], actb[L][br]);
                conv1x1_wmma<256><<<g1, 256, 0, stream>>>(
                    actb[L][br], w2pk[L][br], P(L, br, 6), hout[L][br], Nout);
                break;
            default:
                conv3x3_bn_silu_wmma<512, 20><<<g3, 256, 0, stream>>>(
                    featbf[L], w1pk[L][br], zrow, bna[L][br], bnb[L][br], actb[L][br]);
                conv1x1_wmma<512><<<g1, 256, 0, stream>>>(
                    actb[L][br], w2pk[L][br], P(L, br, 6), hout[L][br], Nout);
                break;
            }
        }
    }

    // ---- decode + sort + NMS, writing d_out ----
    float* o = (float*)d_out;
    size_t obase = 0;
    for (int L = 0; L < 3; ++L) {
        const int H = Ss[L];
        const int N = H * H;            // positions per image
        const int M = BATCH_N * N;

        decode_kernel<<<(M + 255) / 256, 256, 0, stream>>>(
            hout[L][0], hout[L][1], boxesU[L], scoresU[L], labelsU[L], validU[L],
            N, H, Sv[L], M);

        float* oBoxes  = o + obase;
        float* oScores = oBoxes + (size_t)M * 4;
        float* oLabels = oScores + M;
        float* oKeep   = oLabels + M;
        obase += (size_t)M * 7;

        dim3 gs((N + 255) / 256, BATCH_N);
        argsort_scatter_kernel<<<gs, 256, 0, stream>>>(
            boxesU[L], scoresU[L], labelsU[L], validU[L],
            oBoxes, oScores, oLabels, sLabels[L], sValid[L], N);

        nms_kernel<<<BATCH_N, 256, 0, stream>>>(
            oBoxes, sLabels[L], sValid[L], oKeep, N);
    }
}
